// LLMSelector_47931835023417
// MI455X (gfx1250) — compile-verified, matrix operand order checked by reading, not run on hardware
//
#include <hip/hip_runtime.h>
#include <math.h>

typedef _Float16 v16h __attribute__((ext_vector_type(16)));
typedef _Float16 h8   __attribute__((ext_vector_type(8)));
typedef float    v8f  __attribute__((ext_vector_type(8)));

#define DIM    384
#define TWO_D  768
#define LCNT   64
#define HDIM   64
#define RCNT   8

// ---------------------------------------------------------------------------
// Kernel 1: gate logits, softmax, top-2, gate weights, aux-loss partial sums.
// One wave (32 lanes) per row n.
// ---------------------------------------------------------------------------
__global__ __launch_bounds__(256) void gate_kernel(
    const float* __restrict__ posts, const float* __restrict__ reason,
    const float* __restrict__ gate_w, const float* __restrict__ gate_b,
    int* __restrict__ topk, float* __restrict__ gatew,
    float* __restrict__ auxsums)
{
    __shared__ float bsum[16];
    const int tid  = threadIdx.x;
    const int wave = tid >> 5;
    const int lane = tid & 31;
    if (tid < 16) bsum[tid] = 0.f;
    __syncthreads();

    const int n = blockIdx.x * 8 + wave;
    float acc[8];
    #pragma unroll
    for (int r = 0; r < 8; ++r) acc[r] = 0.f;

    for (int d = lane; d < TWO_D; d += 32) {
        float xv = (d < DIM) ? posts[(size_t)n * DIM + d]
                             : reason[(size_t)n * DIM + d - DIM];
        const float4* g = (const float4*)(gate_w + (size_t)d * 8);
        float4 g0 = g[0], g1 = g[1];
        acc[0] += xv * g0.x; acc[1] += xv * g0.y;
        acc[2] += xv * g0.z; acc[3] += xv * g0.w;
        acc[4] += xv * g1.x; acc[5] += xv * g1.y;
        acc[6] += xv * g1.z; acc[7] += xv * g1.w;
    }
    #pragma unroll
    for (int r = 0; r < 8; ++r)
        #pragma unroll
        for (int m = 16; m >= 1; m >>= 1)
            acc[r] += __shfl_xor(acc[r], m, 32);

    if (lane == 0) {
        float lg[8], p[8];
        #pragma unroll
        for (int r = 0; r < 8; ++r) lg[r] = acc[r] + gate_b[r];
        float mx = lg[0];
        #pragma unroll
        for (int r = 1; r < 8; ++r) mx = fmaxf(mx, lg[r]);
        float s = 0.f;
        #pragma unroll
        for (int r = 0; r < 8; ++r) { p[r] = expf(lg[r] - mx); s += p[r]; }
        #pragma unroll
        for (int r = 0; r < 8; ++r) p[r] /= s;

        // top-2 (first occurrence wins on ties, matching lax.top_k)
        int i0 = 0; float v0 = lg[0];
        #pragma unroll
        for (int r = 1; r < 8; ++r) if (lg[r] > v0) { v0 = lg[r]; i0 = r; }
        int i1 = -1; float v1 = -3.4e38f;
        #pragma unroll
        for (int r = 0; r < 8; ++r)
            if (r != i0 && lg[r] > v1) { v1 = lg[r]; i1 = r; }

        float gw0 = 1.f / (1.f + expf(v1 - v0));   // softmax over [v0, v1]
        topk[2 * n]     = i0;   topk[2 * n + 1]  = i1;
        gatew[2 * n]    = gw0;  gatew[2 * n + 1] = 1.f - gw0;

        #pragma unroll
        for (int r = 0; r < 8; ++r) atomicAdd(&bsum[r], p[r]);
        atomicAdd(&bsum[8 + i0], 1.f);
        atomicAdd(&bsum[8 + i1], 1.f);
    }
    __syncthreads();
    if (tid < 16) atomicAdd(&auxsums[tid], bsum[tid]);
}

// ---------------------------------------------------------------------------
// Kernel 2: Vl[r][l][h] = f16( l2norm_h( llm[l] @ V_w[r] + V_b[r] ) )
// One 64-thread block per (r, l); thread = h.
// ---------------------------------------------------------------------------
__global__ __launch_bounds__(64) void vproj_kernel(
    const float* __restrict__ llm, const float* __restrict__ V_w,
    const float* __restrict__ V_b, _Float16* __restrict__ Vl)
{
    const int b = blockIdx.x;
    const int r = b >> 6;
    const int l = b & 63;
    const int h = threadIdx.x;

    float acc = V_b[r * HDIM + h];
    const float* w  = V_w + (size_t)r * DIM * HDIM + h;
    const float* xr = llm + (size_t)l * DIM;
    for (int d = 0; d < DIM; ++d) acc += xr[d] * w[(size_t)d * HDIM];

    __shared__ float red[64];
    red[h] = acc * acc;
    __syncthreads();
    for (int s = 32; s >= 1; s >>= 1) {
        if (h < s) red[h] += red[h + s];
        __syncthreads();
    }
    float inv = 1.f / fmaxf(sqrtf(red[0]), 1e-12f);
    Vl[((size_t)(r * LCNT + l)) * HDIM + h] = (_Float16)(acc * inv);
}

// ---------------------------------------------------------------------------
// Kernel 3: Uh[r][h][d] = f16( U_w[r][d][h] )  (K-contiguous for B fragments)
// ---------------------------------------------------------------------------
__global__ __launch_bounds__(256) void uconv_kernel(
    const float* __restrict__ U_w, _Float16* __restrict__ Uh)
{
    const int o = blockIdx.x * 256 + threadIdx.x;     // RCNT*HDIM*TWO_D = 393216
    if (o >= RCNT * HDIM * TWO_D) return;
    const int d  = o % TWO_D;
    const int rh = o / TWO_D;
    const int h  = rh % HDIM;
    const int r  = rh / HDIM;
    Uh[o] = (_Float16)U_w[((size_t)r * TWO_D + d) * HDIM + h];
}

// ---------------------------------------------------------------------------
// Kernel 4: fused WMMA pipeline. 4 waves/WG, 16 rows/wave, 64 rows/WG.
// ---------------------------------------------------------------------------
#define XS 776   // LDS x-tile stride (halfs)
#define US 72    // per-wave u-stage stride (halfs)
#define FS 68    // per-wave llm-prob stride (floats)

__global__ __launch_bounds__(128) void moe_main_kernel(
    const float* __restrict__ posts, const float* __restrict__ reason,
    const float* __restrict__ rnd,
    const _Float16* __restrict__ Uh, const float* __restrict__ U_b,
    const _Float16* __restrict__ Vl,
    const int* __restrict__ topk, const float* __restrict__ gatew,
    float* __restrict__ out_sel, float* __restrict__ out_logp)
{
    extern __shared__ unsigned char smem[];
    _Float16* xt = (_Float16*)smem;                       // [64][XS] f16
    const int tid  = threadIdx.x;
    const int wave = tid >> 5;
    const int lane = tid & 31;
    const int rowBase = blockIdx.x * 64;

    // ---- stage x tile (f32 -> f16) into LDS, read x exactly once ----
    for (int idx = tid; idx < 64 * 192; idx += 128) {
        int row = idx / 192, seg = idx % 192;
        int n = rowBase + row;
        const float* src = (seg < 96) ? (posts  + (size_t)n * DIM + seg * 4)
                                      : (reason + (size_t)n * DIM + (seg - 96) * 4);
        float4 v = *(const float4*)src;
        _Float16* dst = xt + row * XS + seg * 4;
        dst[0] = (_Float16)v.x; dst[1] = (_Float16)v.y;
        dst[2] = (_Float16)v.z; dst[3] = (_Float16)v.w;
    }
    __syncthreads();

    unsigned char* wbase = smem + 64 * XS * 2 + wave * (16 * FS * 4);
    float*    wscr_f = (float*)wbase;                     // [16][FS] f32 view
    _Float16* wscr_h = (_Float16*)wbase;                  // [16][US] f16 view

    const int hs    = (lane >> 4) & 1;                    // half-wave select
    const int cn    = lane & 15;                          // N/col index
    const int wrow0 = wave * 16;

    // gating info for the 8 rows this lane touches (M = i + 8*hs)
    int   i0v[8], i1v[8];
    float g0v[8], g1v[8];
    #pragma unroll
    for (int i = 0; i < 8; ++i) {
        int n = rowBase + wrow0 + i + 8 * hs;
        i0v[i] = topk[2 * n];  i1v[i] = topk[2 * n + 1];
        g0v[i] = gatew[2 * n]; g1v[i] = gatew[2 * n + 1];
    }

    float llmacc[4][8];
    #pragma unroll
    for (int t = 0; t < 4; ++t)
        #pragma unroll
        for (int i = 0; i < 8; ++i) llmacc[t][i] = 0.f;

    const v8f zz = {0.f, 0.f, 0.f, 0.f, 0.f, 0.f, 0.f, 0.f};

    for (int r = 0; r < RCNT; ++r) {
        // ---- GEMM1: Ux tile [16 x 64] = x_tile[16 x 768] @ U_w[r] ----
        v8f acc[4];
        #pragma unroll
        for (int t = 0; t < 4; ++t) acc[t] = zz;

        const _Float16* urow = Uh + (size_t)r * HDIM * TWO_D;
        for (int kc = 0; kc < TWO_D; kc += 32) {
            const _Float16* ap = xt + (wrow0 + cn) * XS + kc + hs * 8;
            h8 alo = *(const h8*)ap;
            h8 ahi = *(const h8*)(ap + 16);
            v16h a;
            #pragma unroll
            for (int j = 0; j < 8; ++j) { a[j] = alo[j]; a[j + 8] = ahi[j]; }
            #pragma unroll
            for (int t = 0; t < 4; ++t) {
                const _Float16* bp = urow + (size_t)(t * 16 + cn) * TWO_D + kc + hs * 8;
                h8 blo = *(const h8*)bp;
                h8 bhi = *(const h8*)(bp + 16);
                v16h b;
                #pragma unroll
                for (int j = 0; j < 8; ++j) { b[j] = blo[j]; b[j + 8] = bhi[j]; }
                acc[t] = __builtin_amdgcn_wmma_f32_16x16x32_f16(
                    false, a, false, b, (short)0, acc[t], false, false);
            }
        }

        // ---- bias + per-row L2 norm (half-wave shuffle reduce over h) ----
        float ub[4];
        #pragma unroll
        for (int t = 0; t < 4; ++t) ub[t] = U_b[r * HDIM + t * 16 + cn];

        float inv[8];
        #pragma unroll
        for (int i = 0; i < 8; ++i) {
            float s = 0.f;
            #pragma unroll
            for (int t = 0; t < 4; ++t) {
                float v = acc[t][i] + ub[t];
                acc[t][i] = v;
                s += v * v;
            }
            #pragma unroll
            for (int m = 1; m < 16; m <<= 1) s += __shfl_xor(s, m, 16);
            inv[i] = 1.f / fmaxf(sqrtf(s), 1e-12f);
        }

        // ---- restage normalized u into A-fragment layout (f16 in LDS) ----
        #pragma unroll
        for (int i = 0; i < 8; ++i)
            #pragma unroll
            for (int t = 0; t < 4; ++t)
                wscr_h[(i + 8 * hs) * US + t * 16 + cn] =
                    (_Float16)(acc[t][i] * inv[i]);
        asm volatile("s_wait_dscnt 0" ::: "memory");

        // ---- GEMM2: scores [16 x 64] = u[16 x 64] @ Vl[r]^T ----
        v8f sc[4];
        #pragma unroll
        for (int t = 0; t < 4; ++t) sc[t] = zz;
        const _Float16* vrow = Vl + (size_t)r * LCNT * HDIM;
        #pragma unroll
        for (int kc = 0; kc < HDIM; kc += 32) {
            const _Float16* ap = wscr_h + cn * US + kc + hs * 8;
            h8 alo = *(const h8*)ap;
            h8 ahi = *(const h8*)(ap + 16);
            v16h a;
            #pragma unroll
            for (int j = 0; j < 8; ++j) { a[j] = alo[j]; a[j + 8] = ahi[j]; }
            #pragma unroll
            for (int t = 0; t < 4; ++t) {
                const _Float16* bp = vrow + (size_t)(t * 16 + cn) * HDIM + kc + hs * 8;
                h8 blo = *(const h8*)bp;
                h8 bhi = *(const h8*)(bp + 16);
                v16h b;
                #pragma unroll
                for (int j = 0; j < 8; ++j) { b[j] = blo[j]; b[j + 8] = bhi[j]; }
                sc[t] = __builtin_amdgcn_wmma_f32_16x16x32_f16(
                    false, a, false, b, (short)0, sc[t], false, false);
            }
        }

        // ---- softmax over l + gated accumulation into llm_probs ----
        #pragma unroll
        for (int i = 0; i < 8; ++i) {
            float m = sc[0][i];
            #pragma unroll
            for (int t = 1; t < 4; ++t) m = fmaxf(m, sc[t][i]);
            #pragma unroll
            for (int mm = 1; mm < 16; mm <<= 1)
                m = fmaxf(m, __shfl_xor(m, mm, 16));
            float e[4], s = 0.f;
            #pragma unroll
            for (int t = 0; t < 4; ++t) { e[t] = expf(sc[t][i] - m); s += e[t]; }
            #pragma unroll
            for (int mm = 1; mm < 16; mm <<= 1) s += __shfl_xor(s, mm, 16);
            float w = (i0v[i] == r) ? g0v[i] : ((i1v[i] == r) ? g1v[i] : 0.f);
            float wi = w / s;
            #pragma unroll
            for (int t = 0; t < 4; ++t) llmacc[t][i] += e[t] * wi;
        }
    }

    // ---- cumsum / sample / log-prob per row ----
    asm volatile("" ::: "memory");
    #pragma unroll
    for (int i = 0; i < 8; ++i)
        #pragma unroll
        for (int t = 0; t < 4; ++t)
            wscr_f[(i + 8 * hs) * FS + t * 16 + cn] = llmacc[t][i];
    asm volatile("s_wait_dscnt 0" ::: "memory");

    if (lane < 16) {
        int n = rowBase + wrow0 + lane;
        float rv = rnd[n];
        float c = 0.f, val = 0.f;
        float p0 = wscr_f[lane * FS];
        int sel = 0, found = 0;
        for (int l = 0; l < LCNT; ++l) {
            float p = wscr_f[lane * FS + l];
            c += p;
            if (!found && c > rv) { sel = l; val = p; found = 1; }
        }
        if (!found) { sel = 0; val = p0; }
        out_sel[n]  = (float)sel;
        out_logp[n] = logf(val);
    }
}

// ---------------------------------------------------------------------------
// Kernel 5: finalize aux loss.
// ---------------------------------------------------------------------------
__global__ void aux_kernel(const float* __restrict__ auxsums,
                           float* __restrict__ out, float invN)
{
    if (threadIdx.x == 0) {
        float s = 0.f;
        #pragma unroll
        for (int r = 0; r < 8; ++r)
            s += (auxsums[r] * invN) * (auxsums[8 + r] * invN);
        out[0] = 8.f * s * 0.05f;
    }
}

// ---------------------------------------------------------------------------
extern "C" void kernel_launch(void* const* d_in, const int* in_sizes, int n_in,
                              void* d_out, int out_size, void* d_ws, size_t ws_size,
                              hipStream_t stream)
{
    const float* posts  = (const float*)d_in[0];
    const float* reason = (const float*)d_in[1];
    const float* llm    = (const float*)d_in[2];
    const float* rnd    = (const float*)d_in[3];
    const float* gate_w = (const float*)d_in[4];
    const float* gate_b = (const float*)d_in[5];
    const float* U_w    = (const float*)d_in[6];
    const float* U_b    = (const float*)d_in[7];
    const float* V_w    = (const float*)d_in[8];
    const float* V_b    = (const float*)d_in[9];

    const int N = in_sizes[3];          // random_num has N elements

    // workspace layout (16B-aligned sections)
    unsigned char* ws = (unsigned char*)d_ws;
    float*    auxsums = (float*)ws;                                  // 16 f32
    int*      topk    = (int*)(ws + 64);                             // 2N i32
    float*    gatew   = (float*)(ws + 64 + (size_t)2 * N * 4);       // 2N f32
    _Float16* Vl      = (_Float16*)(ws + 64 + (size_t)4 * N * 4);    // R*L*H f16
    _Float16* Uh      = (_Float16*)(ws + 64 + (size_t)4 * N * 4
                                    + (size_t)RCNT * LCNT * HDIM * 2); // R*H*768 f16

    float* out_sel  = (float*)d_out;
    float* out_logp = out_sel + N;
    float* out_aux  = out_sel + (size_t)2 * N;

    hipMemsetAsync(auxsums, 0, 64, stream);

    gate_kernel<<<N / 8, 256, 0, stream>>>(posts, reason, gate_w, gate_b,
                                           topk, gatew, auxsums);
    vproj_kernel<<<RCNT * LCNT, 64, 0, stream>>>(llm, V_w, V_b, Vl);
    uconv_kernel<<<(RCNT * HDIM * TWO_D + 255) / 256, 256, 0, stream>>>(U_w, Uh);

    const size_t smem = (size_t)64 * XS * 2 + (size_t)4 * 16 * FS * 4; // 116,736 B
    (void)hipFuncSetAttribute((const void*)moe_main_kernel,
                              hipFuncAttributeMaxDynamicSharedMemorySize,
                              (int)smem);
    moe_main_kernel<<<N / 64, 128, smem, stream>>>(posts, reason, rnd,
                                                   Uh, U_b, Vl, topk, gatew,
                                                   out_sel, out_logp);

    aux_kernel<<<1, 32, 0, stream>>>(auxsums, out_aux, 1.f / (float)N);
}
